// LocalFeatureTransformer_48670569398602
// MI455X (gfx1250) — matrix operand order repaired and checked by reading, not compiled, not run on hardware
//
#include <hip/hip_runtime.h>
#include <math.h>

typedef _Float16 half_t;
typedef __attribute__((ext_vector_type(16))) _Float16 v16h;
typedef __attribute__((ext_vector_type(8)))  _Float16 v8h;
typedef __attribute__((ext_vector_type(8)))  float    v8f;

#define NB      2
#define HH      96
#define WWI     96
#define LL      (HH*WWI)        // 9216
#define CC      256
#define NWIN    144
#define WGX     12
#define WT      64
#define WBATCH  (NB*NWIN)       // 288
#define TQ      (WBATCH*WT)     // 18432
#define KSEL    8
#define NSEL    (NWIN-KSEL)     // 136
#define SL      (KSEL*WT+NSEL)  // 648
#define TS      (WBATCH*SL)     // 186624
#define NH      8
#define HD      32

__device__ inline float elu1(float x){ return x > 0.f ? x + 1.f : __expf(x); }

// ---------------- WMMA fragment loaders (A: MxK row-major, B: NxK row-major = B^T) ----
__device__ inline v16h load_a_frag(const half_t* __restrict__ base, int ld, int row, int k0, int lane){
    const half_t* pp = base + (size_t)row*ld + k0 + ((lane>>4)<<3);
    v8h lo = *(const v8h*)pp;
    v8h hi = *(const v8h*)(pp + 16);
    v16h f;
#pragma unroll
    for (int j=0;j<8;++j){ f[j]=lo[j]; f[j+8]=hi[j]; }
    return f;
}
__device__ inline v16h load_b_frag(const half_t* __restrict__ base, int ld, int row, int k0, int lane){
    const half_t* pp = base + (size_t)row*ld + k0 + ((lane>>4)<<3);
    v8h lo = *(const v8h*)pp;
    v8h hi = *(const v8h*)(pp + 16);
    v16h f;
#pragma unroll
    for (int j=0;j<8;++j){ f[j]=lo[j]; f[j+8]=hi[j]; }
    return f;
}

// ---------------- Generic WMMA GEMM: C[M x N] = A[M x K] * W[N x K]^T ------------------
// Each wave computes a 16 x (16*NT) strip: the A fragment is loaded once per K-chunk and
// reused across NT WMMAs, cutting A-side DRAM/L2 traffic by NT (the big projections are
// A-traffic bound: 95MB A x 16 col-tiles = 1.5GB without blocking, 375MB with NT=4).
template<int NT>
__global__ void k_gemm_wmma(const half_t* __restrict__ Ag, int lda,
                            const half_t* __restrict__ Wg, int ldw,
                            float* Cf, half_t* Ch, int ldc,
                            int M, int N, int Kd,
                            long long sA, long long sW, long long sC, int relu)
{
    int wave = threadIdx.x >> 5;
    int lane = threadIdx.x & 31;
    int ngrp = (N >> 4) / NT;                    // groups of NT column-tiles
    int tile = blockIdx.x * (blockDim.x >> 5) + wave;
    int total = (M >> 4) * ngrp;
    if (tile >= total) return;                   // wave-uniform: EXEC stays all-ones
    int mt = tile / ngrp, ng = tile % ngrp;
    int bz = blockIdx.y;
    const half_t* A = Ag + (size_t)bz * sA;
    const half_t* W = Wg + (size_t)bz * sW;
    int m0 = mt << 4, n0 = (ng * NT) << 4;
    int mrow = m0 + (lane & 15);
    v8f acc[NT];
#pragma unroll
    for (int t=0;t<NT;++t) acc[t] = (v8f){0.f,0.f,0.f,0.f,0.f,0.f,0.f,0.f};
    for (int k0 = 0; k0 < Kd; k0 += 32){
        if (k0 + 64 <= Kd)   // stream the next A K-slab toward the WGP (global_prefetch_b8)
            __builtin_prefetch((const void*)(A + (size_t)mrow*lda + k0 + 64), 0, 1);
        v16h af = load_a_frag(A, lda, mrow, k0, lane);
#pragma unroll
        for (int t=0;t<NT;++t){
            int nrow = n0 + (t << 4) + (lane & 15);
            v16h bf = load_b_frag(W, ldw, nrow, k0, lane);
            acc[t] = __builtin_amdgcn_wmma_f32_16x16x32_f16(false, af, false, bf,
                                                            (short)0, acc[t], false, false);
        }
    }
    int rbase = m0 + ((lane >> 4) << 3);
#pragma unroll
    for (int t=0;t<NT;++t){
        int col = n0 + (t << 4) + (lane & 15);
#pragma unroll
        for (int v = 0; v < 8; ++v){
            float x = acc[t][v];
            if (relu && x < 0.f) x = 0.f;
            size_t off = (size_t)bz * sC + (size_t)(rbase + v) * ldc + col;
            if (Cf) Cf[off] = x;
            if (Ch) Ch[off] = (half_t)x;
        }
    }
}

// ---------------- window split (pad_split) + f16 copy + mask --------------------------
__global__ void k_winsplit(const float* __restrict__ feat, const unsigned char* __restrict__ mask,
                           float* __restrict__ fw, half_t* __restrict__ fwh, float* __restrict__ mwq)
{
    size_t idx = (size_t)blockIdx.x * blockDim.x + threadIdx.x;
    if (idx >= (size_t)TQ * CC) return;
    int c = idx & 255;
    size_t tok = idx >> 8;
    int b = (int)(tok / (NWIN*WT));
    int r = (int)(tok % (NWIN*WT));
    int win = r >> 6, pos = r & 63;
    int y = (win / WGX) * 8 + (pos >> 3);
    int x = (win % WGX) * 8 + (pos & 7);
    size_t l = (size_t)b * LL + (size_t)y * WWI + x;
    float v = feat[l * CC + c];
    fw[idx] = v; fwh[idx] = (half_t)v;
    if (c == 0) mwq[tok] = mask[l] ? 1.f : 0.f;
}

// ---------------- softmax over 16 ch, avg over 64 tokens, rep = avg@conv_w ------------
__global__ void k_winstats(const float* __restrict__ sim, const float* __restrict__ convw,
                           const float* __restrict__ mwq,
                           float* __restrict__ rep, half_t* __restrict__ reph,
                           half_t* __restrict__ repnh, float* __restrict__ mrep)
{
    int wb = blockIdx.x;
    int t  = threadIdx.x;          // 64 threads
    __shared__ float sm[WT*16];
    __shared__ float avg[16];
    __shared__ float red[WT];
    __shared__ float nrm;
    {
        const float* sp = sim + ((size_t)wb*WT + t) * 16;
        float v[16]; float mx = -3.0e38f;
#pragma unroll
        for (int j=0;j<16;++j){ v[j]=sp[j]; mx = v[j] > mx ? v[j] : mx; }
        float s = 0.f;
#pragma unroll
        for (int j=0;j<16;++j){ v[j]=__expf(v[j]-mx); s += v[j]; }
        float inv = 1.f/s;
#pragma unroll
        for (int j=0;j<16;++j) sm[t*16+j] = v[j]*inv;
    }
    __syncthreads();
    if (t < 16){
        float a = 0.f;
        for (int i=0;i<WT;++i) a += sm[i*16 + t];
        avg[t] = a * (1.f/WT);
    }
    if (t == 0){
        float any = 0.f;
        for (int i=0;i<WT;++i) if (mwq[(size_t)wb*WT+i] > 0.f) any = 1.f;
        mrep[wb] = any;
    }
    __syncthreads();
    float vals[4]; float ssq = 0.f;
#pragma unroll
    for (int r2=0;r2<4;++r2){
        int c = t + WT*r2;
        float a = 0.f;
#pragma unroll
        for (int j=0;j<16;++j) a += avg[j] * convw[j*CC + c];
        vals[r2] = a; ssq += a*a;
        rep[(size_t)wb*CC + c]  = a;
        reph[(size_t)wb*CC + c] = (half_t)a;
    }
    red[t] = ssq; __syncthreads();
    if (t == 0){
        float s = 0.f; for (int i=0;i<WT;++i) s += red[i];
        float n = sqrtf(s); nrm = n < 1e-8f ? 1e-8f : n;
    }
    __syncthreads();
    float inv = 1.f / nrm;
#pragma unroll
    for (int r2=0;r2<4;++r2){
        int c = t + WT*r2;
        repnh[(size_t)wb*CC + c] = (half_t)(vals[r2]*inv);
    }
}

// ---------------- top-K selection + remaining indices ---------------------------------
__global__ void k_topk(const float* __restrict__ sim, int transposed,
                       int* __restrict__ sel, int* __restrict__ ns)
{
    int row = blockIdx.x * blockDim.x + threadIdx.x;
    if (row >= WBATCH) return;
    int b = row / NWIN, r = row % NWIN;
    const float* S = sim + (size_t)b * NWIN * NWIN;
    bool chosen[NWIN] = {false};
    for (int kk=0; kk<KSEL; ++kk){
        float best = -3.0e38f; int bi = 0;
        for (int j=0;j<NWIN;++j){
            if (chosen[j]) continue;
            float vv = transposed ? S[j*NWIN + r] : S[r*NWIN + j];
            if (vv > best){ best = vv; bi = j; }
        }
        chosen[bi] = true;
        sel[row*KSEL + kk] = bi;
    }
    int c = 0;
    for (int j=0;j<NWIN;++j) if (!chosen[j]) ns[row*NSEL + (c++)] = j;
}

// ---------------- gather source t (648 tokens per window) + its mask ------------------
__global__ void k_gather(const half_t* __restrict__ fwh_src, const half_t* __restrict__ reph_src,
                         const float* __restrict__ mw_src, const float* __restrict__ mrep_src,
                         const int* __restrict__ sel, const int* __restrict__ ns,
                         half_t* __restrict__ th, float* __restrict__ mt)
{
    size_t idx = (size_t)blockIdx.x * blockDim.x + threadIdx.x;
    if (idx >= (size_t)TS * CC) return;
    int c = idx & 255;
    size_t r = idx >> 8;
    int wb = (int)(r / SL);
    int j  = (int)(r % SL);
    int b  = wb / NWIN;
    half_t val; float m;
    if (j < KSEL*WT){
        int w = sel[wb*KSEL + (j >> 6)];
        size_t tok = (size_t)(b*NWIN + w)*WT + (j & 63);
        val = fwh_src[tok*CC + c];
        m   = mw_src[tok];
    } else {
        int w = ns[wb*NSEL + (j - KSEL*WT)];
        val = reph_src[(size_t)(b*NWIN + w)*CC + c];
        m   = mrep_src[b*NWIN + w];
    }
    th[idx] = val;
    if (c == 0) mt[r] = m;
}

// ---------------- Ksum[wb][c] = sum_s (elu(k)+1)*mask --------------------------------
__global__ void k_ksum(const half_t* __restrict__ kh, const float* __restrict__ mt,
                       float* __restrict__ ksum)
{
    int wb = blockIdx.x, c = threadIdx.x;
    const half_t* kp = kh + (size_t)wb*SL*CC + c;
    const float*  mp = mt + (size_t)wb*SL;
    float s = 0.f;
    for (int j=0;j<SL;++j){
        float kx = (float)kp[(size_t)j*CC];
        s += elu1(kx) * mp[j];
    }
    ksum[(size_t)wb*CC + c] = s;
}

// ---------------- KV[wb][h] (32x32) = Kf^T * (v/SL), LDS-staged WMMA ------------------
__global__ void k_kv(const half_t* __restrict__ kh, const half_t* __restrict__ vh,
                     const float* __restrict__ mt, float* __restrict__ kvout)
{
    int wb = blockIdx.x, h = blockIdx.y;
    __shared__ half_t kfT[32*32];   // [d][s]
    __shared__ half_t vT [32*32];   // [e][s]
    int tid = threadIdx.x;          // 128 threads = 4 waves
    int wave = tid >> 5, lane = tid & 31;
    int dt = wave >> 1, et = wave & 1;
    v8f acc = {0.f,0.f,0.f,0.f,0.f,0.f,0.f,0.f};
    for (int s0 = 0; s0 < SL; s0 += 32){
        for (int e = tid; e < 1024; e += 128){
            int d = e >> 5, s = e & 31; int sg = s0 + s;
            float kf = 0.f, vv = 0.f;
            if (sg < SL){
                float m  = mt[(size_t)wb*SL + sg];
                float kx = (float)kh[((size_t)wb*SL + sg)*CC + h*HD + d];
                float vx = (float)vh[((size_t)wb*SL + sg)*CC + h*HD + d];
                kf = elu1(kx) * m;
                vv = vx * m * (1.f/(float)SL);
            }
            kfT[d*32 + s] = (half_t)kf;
            vT [d*32 + s] = (half_t)vv;
        }
        __syncthreads();
        int m = dt*16 + (lane & 15);
        int n = et*16 + (lane & 15);
        v16h af = load_a_frag(kfT, 32, m, 0, lane);
        v16h bf = load_b_frag(vT , 32, n, 0, lane);
        acc = __builtin_amdgcn_wmma_f32_16x16x32_f16(false, af, false, bf,
                                                     (short)0, acc, false, false);
        __syncthreads();
    }
    int d0 = dt*16 + ((lane >> 4) << 3);
    int e  = et*16 + (lane & 15);
    float* out = kvout + ((size_t)wb*NH + h)*HD*HD;
#pragma unroll
    for (int v=0; v<8; ++v) out[(d0+v)*HD + e] = acc[v];
}

// ---------------- Z[tok][h] = 1/(Qf . Ksum + eps) ------------------------------------
__global__ void k_z(const float* __restrict__ q, const float* __restrict__ mwq,
                    const float* __restrict__ ksum, float* __restrict__ z)
{
    int i = blockIdx.x * blockDim.x + threadIdx.x;
    if (i >= TQ*NH) return;
    int tok = i / NH, h = i % NH;
    int wb = tok / WT;
    float m = mwq[tok];
    const float* qp = q + (size_t)tok*CC + h*HD;
    const float* kp = ksum + (size_t)wb*CC + h*HD;
    float s = 0.f;
    for (int d=0; d<HD; ++d) s += elu1(qp[d]) * m * kp[d];
    z[i] = 1.f / (s + 1e-6f);
}

// ---------------- attn[tok][h*32+e] = (Qf * KV)[l][e] * Z * SL -----------------------
__global__ void k_attn(const float* __restrict__ q, const float* __restrict__ mwq,
                       const float* __restrict__ kv, const float* __restrict__ z,
                       half_t* __restrict__ attnh)
{
    int wb = blockIdx.x, h = blockIdx.y;
    int wave = threadIdx.x >> 5, lane = threadIdx.x & 31;   // 8 waves
    int lt = wave >> 1, et = wave & 1;                      // 4 x 2 tiles over 64x32
    int l0 = lt*16, e0 = et*16;
    int m = l0 + (lane & 15);
    int tokA = wb*WT + m;
    float qm = mwq[tokA];
    const float* qp = q + (size_t)tokA*CC + h*HD;
    int kb = (lane >> 4) << 3;
    v16h af;
#pragma unroll
    for (int j=0;j<8;++j) af[j]   = (half_t)(elu1(qp[kb + j])      * qm);
#pragma unroll
    for (int j=0;j<8;++j) af[8+j] = (half_t)(elu1(qp[kb + 16 + j]) * qm);
    int n = e0 + (lane & 15);
    int kb2 = (lane >> 4) << 3;
    const float* kvp = kv + ((size_t)wb*NH + h)*HD*HD;
    v16h bf;
#pragma unroll
    for (int j=0;j<16;++j) bf[j] = (half_t)kvp[(kb2 + j + (j & 8))*HD + n];
    v8f acc = {0.f,0.f,0.f,0.f,0.f,0.f,0.f,0.f};
    acc = __builtin_amdgcn_wmma_f32_16x16x32_f16(false, af, false, bf,
                                                 (short)0, acc, false, false);
#pragma unroll
    for (int v=0; v<8; ++v){
        int row = l0 + ((lane >> 4) << 3) + v;
        int tok = wb*WT + row;
        attnh[(size_t)tok*CC + h*HD + e0 + (lane & 15)] =
            (half_t)(acc[v] * z[tok*NH + h] * (float)SL);
    }
}

// ---------------- layernorm -> cat_h = [x_h | ln(msg)_h] ------------------------------
__device__ inline float block_sum(float v, float* red, int c){
    red[c] = v; __syncthreads();
    for (int s=128; s>0; s>>=1){ if (c < s) red[c] += red[c+s]; __syncthreads(); }
    float r = red[0]; __syncthreads();
    return r;
}
__global__ void k_ln_cat(const float* __restrict__ x, const float* __restrict__ g,
                         const float* __restrict__ b, const half_t* __restrict__ fwh,
                         half_t* __restrict__ cath)
{
    int row = blockIdx.x, c = threadIdx.x;   // 256 threads
    __shared__ float red[CC];
    float v = x[(size_t)row*CC + c];
    float mu = block_sum(v, red, c) * (1.f/CC);
    float d = v - mu;
    float var = block_sum(d*d, red, c) * (1.f/CC);
    float y = d * rsqrtf(var + 1e-5f) * g[c] + b[c];
    cath[(size_t)row*(2*CC) + c]      = fwh[(size_t)row*CC + c];
    cath[(size_t)row*(2*CC) + CC + c] = (half_t)y;
}

// ---------------- layernorm + residual update of fw/fwh -------------------------------
__global__ void k_ln_res(const float* __restrict__ x, const float* __restrict__ g,
                         const float* __restrict__ b, float* __restrict__ fw,
                         half_t* __restrict__ fwh)
{
    int row = blockIdx.x, c = threadIdx.x;
    __shared__ float red[CC];
    float v = x[(size_t)row*CC + c];
    float mu = block_sum(v, red, c) * (1.f/CC);
    float d = v - mu;
    float var = block_sum(d*d, red, c) * (1.f/CC);
    float y = d * rsqrtf(var + 1e-5f) * g[c] + b[c];
    float o = fw[(size_t)row*CC + c] + y;
    fw[(size_t)row*CC + c]  = o;
    fwh[(size_t)row*CC + c] = (half_t)o;
}

// ---------------- fold back to (N, L, C) output ---------------------------------------
__global__ void k_fold(const float* __restrict__ fw0, const float* __restrict__ fw1,
                       float* __restrict__ out)
{
    size_t idx = (size_t)blockIdx.x * blockDim.x + threadIdx.x;
    size_t per = (size_t)NB * LL * CC;
    if (idx >= 2*per) return;
    int which = (int)(idx / per);
    size_t r = idx % per;
    int c = (int)(r & 255);
    size_t t = r >> 8;
    int b = (int)(t / LL);
    int l = (int)(t % LL);
    int y = l / WWI, x = l % WWI;
    int win = (y >> 3) * WGX + (x >> 3);
    int pos = (y & 7) * 8 + (x & 7);
    const float* src = which ? fw1 : fw0;
    out[idx] = src[((size_t)(b*NWIN + win)*WT + pos)*CC + c];
}

// ---------------- utility: f32 -> f16 --------------------------------------------------
__global__ void k_f32_to_f16(half_t* __restrict__ dst, const float* __restrict__ src, size_t n){
    size_t i = (size_t)blockIdx.x * blockDim.x + threadIdx.x;
    if (i < n) dst[i] = (half_t)src[i];
}

// ======================================================================================
static void gemm(hipStream_t st, const half_t* A, int lda, const half_t* W, int ldw,
                 float* Cf, half_t* Ch, int ldc, int M, int N, int Kd,
                 int batch = 1, long long sA = 0, long long sW = 0, long long sC = 0,
                 int relu = 0)
{
    if ((N & 63) == 0){
        int tiles = (M/16) * (N/64);
        dim3 grid((tiles + 7) / 8, batch);
        k_gemm_wmma<4><<<grid, 256, 0, st>>>(A, lda, W, ldw, Cf, Ch, ldc, M, N, Kd, sA, sW, sC, relu);
    } else {
        int tiles = (M/16) * (N/16);
        dim3 grid((tiles + 7) / 8, batch);
        k_gemm_wmma<1><<<grid, 256, 0, st>>>(A, lda, W, ldw, Cf, Ch, ldc, M, N, Kd, sA, sW, sC, relu);
    }
}

extern "C" void kernel_launch(void* const* d_in, const int* in_sizes, int n_in,
                              void* d_out, int out_size, void* d_ws, size_t ws_size,
                              hipStream_t stream)
{
    const float* feat0 = (const float*)d_in[0];
    const float* feat1 = (const float*)d_in[1];
    const unsigned char* mask0 = (const unsigned char*)d_in[2];
    const unsigned char* mask1 = (const unsigned char*)d_in[3];
    const float* qW   = (const float*)d_in[8];
    const float* kW   = (const float*)d_in[9];
    const float* vW   = (const float*)d_in[10];
    const float* mW   = (const float*)d_in[11];
    const float* w1   = (const float*)d_in[12];
    const float* w2   = (const float*)d_in[13];
    const float* n1g  = (const float*)d_in[14];
    const float* n1b  = (const float*)d_in[15];
    const float* n2g  = (const float*)d_in[16];
    const float* n2b  = (const float*)d_in[17];
    const float* convw= (const float*)d_in[18];

    char* p = (char*)d_ws;
    auto alloc = [&](size_t bytes)->void*{ void* r = (void*)p; p += (bytes + 255) & ~(size_t)255; return r; };

    float*  fw[2];  half_t* fwh[2];  float* mwq[2]; float* mrep[2];
    float*  rep[2]; half_t* reph[2]; half_t* repnh[2];
    for (int i=0;i<2;++i){
        fw[i]   = (float*) alloc((size_t)TQ*CC*4);
        fwh[i]  = (half_t*)alloc((size_t)TQ*CC*2);
        mwq[i]  = (float*) alloc((size_t)TQ*4);
        mrep[i] = (float*) alloc((size_t)WBATCH*4);
        rep[i]  = (float*) alloc((size_t)WBATCH*CC*4);
        reph[i] = (half_t*)alloc((size_t)WBATCH*CC*2);
        repnh[i]= (half_t*)alloc((size_t)WBATCH*CC*2);
    }
    half_t* qWh  = (half_t*)alloc((size_t)2*CC*CC*2);
    half_t* kWh  = (half_t*)alloc((size_t)2*CC*CC*2);
    half_t* vWh  = (half_t*)alloc((size_t)2*CC*CC*2);
    half_t* mWh  = (half_t*)alloc((size_t)2*CC*CC*2);
    half_t* w1h  = (half_t*)alloc((size_t)2*(2*CC)*(2*CC)*2);
    half_t* w2h  = (half_t*)alloc((size_t)2*CC*(2*CC)*2);
    half_t* cvh  = (half_t*)alloc((size_t)16*CC*2);
    float*  simbuf = (float*)alloc((size_t)TQ*16*4);
    float*  simww  = (float*)alloc((size_t)NB*NWIN*NWIN*4);
    int* sel[2]; int* ns[2]; half_t* th[2]; float* mtb[2];
    for (int i=0;i<2;++i){
        sel[i] = (int*)alloc((size_t)WBATCH*KSEL*4);
        ns[i]  = (int*)alloc((size_t)WBATCH*NSEL*4);
        th[i]  = (half_t*)alloc((size_t)TS*CC*2);
        mtb[i] = (float*) alloc((size_t)TS*4);
    }
    half_t* kh   = (half_t*)alloc((size_t)TS*CC*2);
    half_t* vhb  = (half_t*)alloc((size_t)TS*CC*2);
    float*  ksum = (float*) alloc((size_t)WBATCH*CC*4);
    float*  kvb  = (float*) alloc((size_t)WBATCH*NH*HD*HD*4);
    float*  qb   = (float*) alloc((size_t)TQ*CC*4);
    float*  zb   = (float*) alloc((size_t)TQ*NH*4);
    half_t* attnh= (half_t*)alloc((size_t)TQ*CC*2);
    float*  msg  = (float*) alloc((size_t)TQ*CC*4);
    half_t* cath = (half_t*)alloc((size_t)TQ*2*CC*2);
    half_t* hh   = (half_t*)alloc((size_t)TQ*2*CC*2);

    // weights -> f16 (once per call)
    auto cvt = [&](half_t* dst, const float* src, size_t n){
        k_f32_to_f16<<<(unsigned)((n + 255)/256), 256, 0, stream>>>(dst, src, n);
    };
    cvt(qWh, qW, (size_t)2*CC*CC);   cvt(kWh, kW, (size_t)2*CC*CC);
    cvt(vWh, vW, (size_t)2*CC*CC);   cvt(mWh, mW, (size_t)2*CC*CC);
    cvt(w1h, w1, (size_t)2*(2*CC)*(2*CC));
    cvt(w2h, w2, (size_t)2*CC*(2*CC));
    cvt(cvh, convw, (size_t)16*CC);

    // window split of the inputs
    {
        size_t tot = (size_t)TQ*CC;
        k_winsplit<<<(unsigned)((tot+255)/256),256,0,stream>>>(feat0, mask0, fw[0], fwh[0], mwq[0]);
        k_winsplit<<<(unsigned)((tot+255)/256),256,0,stream>>>(feat1, mask1, fw[1], fwh[1], mwq[1]);
    }

    for (int layer = 0; layer < 2; ++layer){
        const half_t* qWl = qWh + (size_t)layer*CC*CC;
        const half_t* kWl = kWh + (size_t)layer*CC*CC;
        const half_t* vWl = vWh + (size_t)layer*CC*CC;
        const half_t* mWl = mWh + (size_t)layer*CC*CC;
        const half_t* w1l = w1h + (size_t)layer*(2*CC)*(2*CC);
        const half_t* w2l = w2h + (size_t)layer*CC*(2*CC);

        // preprocess both images
        for (int i=0;i<2;++i){
            gemm(stream, fwh[i], CC, cvh, CC, simbuf, nullptr, 16, TQ, 16, CC);
            k_winstats<<<WBATCH, 64, 0, stream>>>(simbuf, convw, mwq[i],
                                                  rep[i], reph[i], repnh[i], mrep[i]);
        }

        int src[2];
        if (layer == 0){     // self: per-image self-similarity
            for (int i=0;i<2;++i){
                gemm(stream, repnh[i], CC, repnh[i], CC, simww, nullptr, NWIN,
                     NWIN, NWIN, CC, NB, (long long)NWIN*CC, (long long)NWIN*CC,
                     (long long)NWIN*NWIN);
                k_topk<<<(WBATCH+63)/64, 64, 0, stream>>>(simww, 0, sel[i], ns[i]);
            }
            src[0] = 0; src[1] = 1;
        } else {             // cross: sim01 rows -> t0 (from image1), cols -> t1 (from image0)
            gemm(stream, repnh[0], CC, repnh[1], CC, simww, nullptr, NWIN,
                 NWIN, NWIN, CC, NB, (long long)NWIN*CC, (long long)NWIN*CC,
                 (long long)NWIN*NWIN);
            k_topk<<<(WBATCH+63)/64, 64, 0, stream>>>(simww, 0, sel[0], ns[0]);
            k_topk<<<(WBATCH+63)/64, 64, 0, stream>>>(simww, 1, sel[1], ns[1]);
            src[0] = 1; src[1] = 0;
        }

        // gather both source tensors BEFORE any residual update (snapshot semantics)
        for (int i=0;i<2;++i){
            int s = src[i];
            size_t tot = (size_t)TS*CC;
            k_gather<<<(unsigned)((tot+255)/256), 256, 0, stream>>>(
                fwh[s], reph[s], mwq[s], mrep[s], sel[i], ns[i], th[i], mtb[i]);
        }

        // encoder per image
        for (int i=0;i<2;++i){
            gemm(stream, th[i], CC, kWl, CC, nullptr, kh,  CC, TS, CC, CC);
            gemm(stream, th[i], CC, vWl, CC, nullptr, vhb, CC, TS, CC, CC);
            k_ksum<<<WBATCH, CC, 0, stream>>>(kh, mtb[i], ksum);
            { dim3 g(WBATCH, NH); k_kv<<<g, 128, 0, stream>>>(kh, vhb, mtb[i], kvb); }
            gemm(stream, fwh[i], CC, qWl, CC, qb, nullptr, CC, TQ, CC, CC);
            k_z<<<(TQ*NH+255)/256, 256, 0, stream>>>(qb, mwq[i], ksum, zb);
            { dim3 g(WBATCH, NH); k_attn<<<g, 256, 0, stream>>>(qb, mwq[i], kvb, zb, attnh); }
            gemm(stream, attnh, CC, mWl, CC, msg, nullptr, CC, TQ, CC, CC);
            k_ln_cat<<<TQ, CC, 0, stream>>>(msg, n1g + layer*CC, n1b + layer*CC, fwh[i], cath);
            gemm(stream, cath, 2*CC, w1l, 2*CC, nullptr, hh, 2*CC, TQ, 2*CC, 2*CC,
                 1, 0, 0, 0, /*relu*/1);
            gemm(stream, hh, 2*CC, w2l, 2*CC, msg, nullptr, CC, TQ, CC, 2*CC);
            k_ln_res<<<TQ, CC, 0, stream>>>(msg, n2g + layer*CC, n2b + layer*CC, fw[i], fwh[i]);
        }
    }

    // fold back both feature maps into the concatenated output
    {
        size_t tot = (size_t)2*NB*LL*CC;
        k_fold<<<(unsigned)((tot+255)/256), 256, 0, stream>>>(fw[0], fw[1], (float*)d_out);
    }
}
